// VolSDFNetworkBG_39247411151288
// MI455X (gfx1250) — matrix operand order repaired and checked by reading, not compiled, not run on hardware
//
#include <hip/hip_runtime.h>

// ---------------------------------------------------------------------------
// Types for CDNA5 WMMA
// ---------------------------------------------------------------------------
typedef __attribute__((ext_vector_type(16))) __bf16 v16bf;
typedef __attribute__((ext_vector_type(8)))  float  v8f;

union FragBF { unsigned int u[8]; v16bf v; };

// bf16 helpers (round-to-nearest-even)
__device__ __forceinline__ unsigned short f2bf(float f) {
    unsigned int u = __float_as_uint(f);
    unsigned int r = (u + 0x7FFFu + ((u >> 16) & 1u)) >> 16;
    return (unsigned short)r;
}
__device__ __forceinline__ float bf2f(unsigned short h) {
    return __uint_as_float(((unsigned int)h) << 16);
}

#define N_RAYS 512
#define NS_FG  192      // 3*64 fused long set (short = first 128)
#define M_FG   (N_RAYS * NS_FG)    // 98304
#define NBG    32
#define M_BG   (N_RAYS * NBG)      // 16384

#define LDSPAD 72       // padded LDS row (64 data + 8), 144B stride = 36 banks

// ---------------------------------------------------------------------------
// Fused WMMA GEMM:  C[M x N] = act(A[M x K] @ B[K x N] + bias)
// A: bf16 row-major (lda = padded K, mult of 64), B: bf16 row-major (ldb = Np).
// Block tile 128x64, 8 waves of 32x32 (2x2 v_wmma_f32_16x16x32_bf16).
// BK=64, double-buffered LDS: 8 WMMAs per barrier.
// ACT: 0=none 1=softplus100 2=relu 3=sigmoid
// SIG:   store bf16 sigmoid(100*z) (softplus' mask for the VJP)
// MUL:   elementwise bf16 multiplier applied after act (applies softplus' mask)
// SPLIT: col0 -> sdfOut (f32), cols 1..256 -> featOut (bf16)
// ---------------------------------------------------------------------------
template <int ACT, bool SPLIT, bool SIG, bool MUL>
__global__ __launch_bounds__(256) void gemm_bf16_kernel(
    const unsigned short* __restrict__ A, int lda,
    const unsigned short* __restrict__ B, int ldb,
    const float* __restrict__ bias, int biasN,
    int M, int K,
    unsigned short* __restrict__ Cout, int ldc,
    unsigned short* __restrict__ SigOut,
    const unsigned short* __restrict__ Mul,
    float* __restrict__ sdfOut,
    unsigned short* __restrict__ featOut)
{
    __shared__ unsigned short As[2][128][LDSPAD];
    __shared__ unsigned short BsT[2][64][LDSPAD];   // transposed: [n][k]

    const int tid  = threadIdx.x;
    const int wave = tid >> 5;
    const int lane = tid & 31;
    const int l16  = lane & 15;
    const int h    = lane >> 4;
    const int mBase = (wave & 3) * 32;
    const int nBase = (wave >> 2) * 32;
    const int blockM = blockIdx.x * 128;
    const int blockN = blockIdx.y * 64;

    auto stageA = [&](int bufI, int k0) {
        int row = tid >> 1;
        int col = (tid & 1) * 32;
        const uint4* src = (const uint4*)(A + (size_t)(blockM + row) * lda + k0 + col);
        uint4 v0 = src[0]; uint4 v1 = src[1]; uint4 v2 = src[2]; uint4 v3 = src[3];
        *(uint4*)&As[bufI][row][col]      = v0;
        *(uint4*)&As[bufI][row][col + 8]  = v1;
        *(uint4*)&As[bufI][row][col + 16] = v2;
        *(uint4*)&As[bufI][row][col + 24] = v3;
    };
    auto stageB = [&](int bufI, int k0) {
        int kr = tid >> 2;            // 64 k-rows, 4 threads each
        int nc = (tid & 3) * 16;      // 16 contiguous n per thread
        union { uint4 q[2]; unsigned short s[16]; } t;
        const uint4* sp = (const uint4*)(B + (size_t)(k0 + kr) * ldb + blockN + nc);
        t.q[0] = sp[0]; t.q[1] = sp[1];
        #pragma unroll
        for (int i = 0; i < 16; ++i) BsT[bufI][nc + i][kr] = t.s[i];
    };

    v8f acc[2][2];
    #pragma unroll
    for (int i = 0; i < 2; ++i)
        #pragma unroll
        for (int j = 0; j < 2; ++j)
            #pragma unroll
            for (int e = 0; e < 8; ++e) acc[i][j][e] = 0.0f;

    stageA(0, 0);
    stageB(0, 0);
    __syncthreads();

    int cur = 0;
    for (int k0 = 0; k0 < K; k0 += 64) {
        int nxt = cur ^ 1;
        if (k0 + 64 < K) {            // prefetch next tile while computing
            stageA(nxt, k0 + 64);
            stageB(nxt, k0 + 64);
        }
        #pragma unroll
        for (int kk = 0; kk < 64; kk += 32) {
            FragBF a[2], b[2];
            #pragma unroll
            for (int i = 0; i < 2; ++i) {
                int m = mBase + i * 16 + l16;
                #pragma unroll
                for (int r = 0; r < 8; ++r) {
                    int kb = 16 * h + 2 * r;
                    a[i].u[r] = *(const unsigned int*)&As[cur][m][kk + kb];
                }
            }
            #pragma unroll
            for (int j = 0; j < 2; ++j) {
                int n = nBase + j * 16 + l16;
                #pragma unroll
                for (int r = 0; r < 8; ++r)
                    b[j].u[r] = *(const unsigned int*)&BsT[cur][n][kk + 16 * h + 2 * r];
            }
            #pragma unroll
            for (int i = 0; i < 2; ++i)
                #pragma unroll
                for (int j = 0; j < 2; ++j)
                    acc[i][j] = __builtin_amdgcn_wmma_f32_16x16x32_bf16(
                        false, a[i].v, false, b[j].v, (short)0, acc[i][j], false, false);
        }
        __syncthreads();
        cur = nxt;
    }

    // ---- fused, branch-light epilogue
    #pragma unroll
    for (int i = 0; i < 2; ++i) {
        #pragma unroll
        for (int j = 0; j < 2; ++j) {
            #pragma unroll
            for (int r = 0; r < 8; ++r) {
                int m = blockM + mBase + i * 16 + r + 8 * h;
                int n = blockN + nBase + j * 16 + l16;
                if (m >= M) continue;
                float v = acc[i][j][r];
                if (bias != nullptr && n < biasN) v += bias[n];
                float outv = v;
                if (ACT == 1) {            // softplus(100x)/100
                    float t = 100.0f * v;
                    outv = (t > 15.0f) ? v : log1pf(__expf(t)) * 0.01f;
                    if (SIG) SigOut[(size_t)m * 256 + n] = f2bf(1.0f / (1.0f + __expf(-t)));
                } else if (ACT == 2) {
                    outv = fmaxf(v, 0.0f);
                } else if (ACT == 3) {
                    outv = 1.0f / (1.0f + __expf(-v));
                }
                if (MUL) outv *= bf2f(Mul[(size_t)m * 256 + n]);
                if (SPLIT) {
                    if (n == 0)        sdfOut[m] = outv;
                    else if (n <= 256) featOut[(size_t)m * 256 + (n - 1)] = f2bf(outv);
                } else {
                    Cout[(size_t)m * ldc + n] = f2bf(outv);
                }
            }
        }
    }
}

// ---------------------------------------------------------------------------
// Small-N GEMM (N<=3): one thread per row, f32 weights (optionally transposed)
// trans: B_eff[k][n] = W[n*wldN + k]  (else W[k*wldN + n])
// ---------------------------------------------------------------------------
__global__ void gemm_smalln_kernel(
    const unsigned short* __restrict__ A, int lda,
    const float* __restrict__ W, int wldN, int trans,
    const float* __restrict__ bias,
    int M, int K, int Nout, int act,
    float* __restrict__ outF)
{
    int m = blockIdx.x * blockDim.x + threadIdx.x;
    if (m >= M) return;
    float accv[3] = {0.f, 0.f, 0.f};
    for (int k = 0; k < K; ++k) {
        float a = bf2f(A[(size_t)m * lda + k]);
        #pragma unroll
        for (int n = 0; n < 3; ++n) {
            if (n < Nout) {
                float w = trans ? W[(size_t)n * wldN + k] : W[(size_t)k * wldN + n];
                accv[n] = fmaf(a, w, accv[n]);
            }
        }
    }
    for (int n = 0; n < Nout; ++n) {
        float v = accv[n] + (bias ? bias[n] : 0.0f);
        if (act == 3) v = 1.0f / (1.0f + __expf(-v));
        outF[(size_t)m * Nout + n] = v;
    }
}

// ---------------------------------------------------------------------------
// Weight pad/convert (optionally transposed) f32 -> bf16
// ---------------------------------------------------------------------------
__global__ void pad_weight_kernel(const float* __restrict__ src, int K, int N,
                                  unsigned short* __restrict__ dst, int Kp, int Np,
                                  int trans)
{
    size_t t = (size_t)blockIdx.x * blockDim.x + threadIdx.x;
    if (t >= (size_t)Kp * Np) return;
    int i = (int)(t / Np), j = (int)(t % Np);
    float v = 0.0f;
    if (!trans) { if (i < K && j < N) v = src[(size_t)i * N + j]; }
    else        { if (i < N && j < K) v = src[(size_t)j * N + i]; }
    dst[t] = f2bf(v);
}

// ---------------------------------------------------------------------------
// Foreground sample points: pts = cam + z * dir ; X0 = bf16 [M x 64] (pad)
// ---------------------------------------------------------------------------
__global__ void fg_pts_kernel(const float* __restrict__ cam, const float* __restrict__ dirs,
                              const float* __restrict__ z1, const float* __restrict__ z2,
                              const float* __restrict__ z3,
                              unsigned short* __restrict__ X0, float* __restrict__ pts)
{
    int idx = blockIdx.x * blockDim.x + threadIdx.x;
    if (idx >= M_FG) return;
    int ray = idx / NS_FG, s = idx % NS_FG;
    float z = (s < 64) ? z1[ray * 64 + s]
            : (s < 128) ? z2[ray * 64 + s - 64]
                        : z3[ray * 64 + s - 128];
    float p[3];
    #pragma unroll
    for (int c = 0; c < 3; ++c) {
        p[c] = cam[ray * 3 + c] + z * dirs[ray * 3 + c];
        pts[(size_t)idx * 3 + c] = p[c];
        X0[(size_t)idx * 64 + c] = f2bf(p[c]);
    }
    for (int c = 3; c < 64; ++c) X0[(size_t)idx * 64 + c] = 0;
}

// g3 = sigmoid(100*z3) * iw3[:,0]
__global__ void g3_kernel(const unsigned short* __restrict__ S3,
                          const float* __restrict__ iw3,
                          unsigned short* __restrict__ out)
{
    size_t t = (size_t)blockIdx.x * blockDim.x + threadIdx.x;
    if (t >= (size_t)M_FG * 256) return;
    int n = (int)(t % 256);
    out[t] = f2bf(bf2f(S3[t]) * iw3[(size_t)n * 257]);
}

// X_r[m, 0:320] = [p(3) d(3) n(3) feat(256) 0-pad]
__global__ void assemble_xr_kernel(const float* __restrict__ pts, const float* __restrict__ dirs,
                                   const float* __restrict__ grads,
                                   const unsigned short* __restrict__ feat,
                                   unsigned short* __restrict__ Xr)
{
    size_t t = (size_t)blockIdx.x * blockDim.x + threadIdx.x;
    if (t >= (size_t)M_FG * 320) return;
    int m = (int)(t / 320), c = (int)(t % 320);
    unsigned short v;
    if (c < 3)        v = f2bf(pts[(size_t)m * 3 + c]);
    else if (c < 6)   v = f2bf(dirs[(m / NS_FG) * 3 + (c - 3)]);
    else if (c < 9)   v = f2bf(grads[(size_t)m * 3 + (c - 6)]);
    else if (c < 265) v = feat[(size_t)m * 256 + (c - 9)];
    else              v = 0;
    Xr[t] = v;
}

// X_g[m, 0:320] = [dir(3) bg_feat(256) 0-pad]
__global__ void assemble_xg_kernel(const float* __restrict__ dirs,
                                   const unsigned short* __restrict__ feat,
                                   unsigned short* __restrict__ Xg)
{
    size_t t = (size_t)blockIdx.x * blockDim.x + threadIdx.x;
    if (t >= (size_t)M_BG * 320) return;
    int m = (int)(t / 320), c = (int)(t % 320);
    unsigned short v;
    if (c < 3)        v = f2bf(dirs[(m / NBG) * 3 + c]);
    else if (c < 259) v = feat[(size_t)m * 256 + (c - 3)];
    else              v = 0;
    Xg[t] = v;
}

// depth2pts_outside geometry for background points (z_bg flipped)
__global__ void bg_pts_kernel(const float* __restrict__ cam, const float* __restrict__ dirs,
                              const float* __restrict__ zbg,
                              unsigned short* __restrict__ Xbg)
{
    int idx = blockIdx.x * blockDim.x + threadIdx.x;
    if (idx >= M_BG) return;
    int ray = idx / NBG, s = idx % NBG;
    float depth = zbg[ray * NBG + (NBG - 1 - s)];
    float ox = cam[ray * 3], oy = cam[ray * 3 + 1], oz = cam[ray * 3 + 2];
    float dx = dirs[ray * 3], dy = dirs[ray * 3 + 1], dz = dirs[ray * 3 + 2];
    float odd = ox * dx + oy * dy + oz * dz;
    float oo  = ox * ox + oy * oy + oz * oz;
    float under = odd * odd - (oo - 9.0f);
    float dsph = sqrtf(fmaxf(under, 1e-6f)) - odd;
    float px = ox + dsph * dx, py = oy + dsph * dy, pz = oz + dsph * dz;
    float mx = ox - odd * dx,  my = oy - odd * dy,  mz = oz - odd * dz;
    float pmn = sqrtf(mx * mx + my * my + mz * mz);
    float rx = oy * pz - oz * py, ry = oz * px - ox * pz, rz = ox * py - oy * px;
    float rn = rsqrtf(rx * rx + ry * ry + rz * rz);
    rx *= rn; ry *= rn; rz *= rn;
    float ang = asinf(pmn / 3.0f) - asinf(pmn * depth);
    float cA = cosf(ang), sA = sinf(ang);
    float cx = ry * pz - rz * py, cy = rz * px - rx * pz, cz = rx * py - ry * px;
    float rdp = rx * px + ry * py + rz * pz;
    float omc = 1.0f - cA;
    float nx = px * cA + cx * sA + rx * rdp * omc;
    float ny = py * cA + cy * sA + ry * rdp * omc;
    float nz = pz * cA + cz * sA + rz * rdp * omc;
    float inv = rsqrtf(nx * nx + ny * ny + nz * nz);
    nx *= inv; ny *= inv; nz *= inv;
    Xbg[(size_t)idx * 64 + 0] = f2bf(nx);
    Xbg[(size_t)idx * 64 + 1] = f2bf(ny);
    Xbg[(size_t)idx * 64 + 2] = f2bf(nz);
    Xbg[(size_t)idx * 64 + 3] = f2bf(depth);
    for (int c = 4; c < 64; ++c) Xbg[(size_t)idx * 64 + c] = 0;
}

__device__ __forceinline__ float laplace_density(float s) {
    float sgn = (s > 0.0f) ? 1.0f : ((s < 0.0f) ? -1.0f : 0.0f);
    return 10.0f * (0.5f + 0.5f * sgn * expm1f(-fabsf(s) * 10.0f));
}

// Per-ray foreground composite (long 192 + short 128 in one pass, mask select)
__global__ void composite_fg_kernel(const float* __restrict__ z1, const float* __restrict__ z2,
                                    const float* __restrict__ z3,
                                    const unsigned char* __restrict__ mask,
                                    const float* __restrict__ sdf, const float* __restrict__ rgb,
                                    float* __restrict__ fgRgb, float* __restrict__ bgT)
{
    int r = blockIdx.x * blockDim.x + threadIdx.x;
    if (r >= N_RAYS) return;
    float zm1 = z1[r * 64 + 63], zm2 = z2[r * 64 + 63], zm3 = z3[r * 64 + 63];
    float zmaxL = zm1 + zm2 + zm3, zmaxS = zm1 + zm2;
    auto zr = [&](int s) -> float {
        return (s < 64)  ? z1[r * 64 + s]
             : (s < 128) ? z2[r * 64 + s - 64] + zm1
                         : z3[r * 64 + s - 128] + zm1 + zm2;
    };
    float Tl = 1.0f, Ts = 1.0f;
    float aL[3] = {0.f, 0.f, 0.f}, aS[3] = {0.f, 0.f, 0.f};
    for (int s = 0; s < NS_FG; ++s) {
        float zc   = zr(s);
        float zn   = (s < NS_FG - 1) ? zr(s + 1) : zmaxL;
        float dens = laplace_density(sdf[r * NS_FG + s]);
        float feL  = (zn - zc) * dens;
        float eL   = __expf(-feL);
        float wL   = (1.0f - eL) * Tl;
        #pragma unroll
        for (int c = 0; c < 3; ++c) aL[c] += wL * rgb[(size_t)(r * NS_FG + s) * 3 + c];
        Tl *= eL;
        if (s < 128) {
            float znS  = (s == 127) ? zmaxS : zr(s + 1);
            float feS  = (znS - zc) * dens;
            float eS   = __expf(-feS);
            float wS   = (1.0f - eS) * Ts;
            #pragma unroll
            for (int c = 0; c < 3; ++c) aS[c] += wS * rgb[(size_t)(r * NS_FG + s) * 3 + c];
            Ts *= eS;
        }
    }
    bool mk = mask[r] != 0;
    #pragma unroll
    for (int c = 0; c < 3; ++c) fgRgb[r * 3 + c] = mk ? aL[c] : aS[c];
    bgT[r] = mk ? Tl : Ts;
}

// Per-ray background composite + final output
__global__ void composite_bg_kernel(const float* __restrict__ zbg,
                                    const float* __restrict__ bgSdf, const float* __restrict__ bgRgb,
                                    const float* __restrict__ fgRgb, const float* __restrict__ bgT,
                                    float* __restrict__ outp)
{
    int r = blockIdx.x * blockDim.x + threadIdx.x;
    if (r >= N_RAYS) return;
    float T = 1.0f;
    float acc[3] = {0.f, 0.f, 0.f};
    for (int s = 0; s < NBG; ++s) {
        float zf   = zbg[r * NBG + (NBG - 1 - s)];
        float dist = (s < NBG - 1) ? (zf - zbg[r * NBG + (NBG - 2 - s)]) : 1e10f;
        float dens = fabsf(bgSdf[r * NBG + s]);
        float bfe  = dist * dens;
        float e    = __expf(-bfe);
        float w    = (1.0f - e) * T;
        #pragma unroll
        for (int c = 0; c < 3; ++c) acc[c] += w * bgRgb[(size_t)(r * NBG + s) * 3 + c];
        T *= e;
    }
    #pragma unroll
    for (int c = 0; c < 3; ++c)
        outp[r * 3 + c] = fgRgb[r * 3 + c] + bgT[r] * acc[c];
}

// ---------------------------------------------------------------------------
// Host launcher
// ---------------------------------------------------------------------------
enum { CFG_SP_SIG = 0, CFG_SPLIT, CFG_MUL, CFG_RELU, CFG_SP };

extern "C" void kernel_launch(void* const* d_in, const int* in_sizes, int n_in,
                              void* d_out, int out_size, void* d_ws, size_t ws_size,
                              hipStream_t stream) {
    (void)in_sizes; (void)n_in; (void)out_size; (void)ws_size;
    const float* cam  = (const float*)d_in[0];
    const float* dirs = (const float*)d_in[1];
    const float* z1   = (const float*)d_in[2];
    const float* z2   = (const float*)d_in[3];
    const float* z3   = (const float*)d_in[4];
    const float* zbg  = (const float*)d_in[5];
    const unsigned char* mask = (const unsigned char*)d_in[6];
    const float* iw0 = (const float*)d_in[7];  const float* ib0 = (const float*)d_in[8];
    const float* iw1 = (const float*)d_in[9];  const float* ib1 = (const float*)d_in[10];
    const float* iw2 = (const float*)d_in[11]; const float* ib2 = (const float*)d_in[12];
    const float* iw3 = (const float*)d_in[13]; const float* ib3 = (const float*)d_in[14];
    const float* rw0 = (const float*)d_in[15]; const float* rb0 = (const float*)d_in[16];
    const float* rw1 = (const float*)d_in[17]; const float* rb1 = (const float*)d_in[18];
    const float* rw2 = (const float*)d_in[19]; const float* rb2 = (const float*)d_in[20];
    const float* bw0 = (const float*)d_in[21]; const float* bb0 = (const float*)d_in[22];
    const float* bw1 = (const float*)d_in[23]; const float* bb1 = (const float*)d_in[24];
    const float* bw2 = (const float*)d_in[25]; const float* bb2 = (const float*)d_in[26];
    const float* gw0 = (const float*)d_in[27]; const float* gb0 = (const float*)d_in[28];
    const float* gw1 = (const float*)d_in[29]; const float* gb1 = (const float*)d_in[30];
    float* outp = (float*)d_out;

    // ---- workspace carve
    char* base = (char*)d_ws; size_t off = 0;
    auto carve = [&](size_t n) -> void* {
        void* p = base + off; off = (off + n + 255) & ~(size_t)255; return p;
    };
    const int M = M_FG, MB = M_BG;
    unsigned short* X0   = (unsigned short*)carve((size_t)M * 64 * 2);
    float*          pts  = (float*)carve((size_t)M * 3 * 4);
    unsigned short* bufA = (unsigned short*)carve((size_t)M * 256 * 2);
    unsigned short* bufB = (unsigned short*)carve((size_t)M * 256 * 2);
    unsigned short* S1   = (unsigned short*)carve((size_t)M * 256 * 2);
    unsigned short* S2   = (unsigned short*)carve((size_t)M * 256 * 2);
    unsigned short* S3   = (unsigned short*)carve((size_t)M * 256 * 2);
    unsigned short* Xr   = (unsigned short*)carve((size_t)M * 320 * 2);
    float*          sdf  = (float*)carve((size_t)M * 4);
    float*          grads= (float*)carve((size_t)M * 3 * 4);
    float*          rgb  = (float*)carve((size_t)M * 3 * 4);
    float*          fgRgb= (float*)carve(N_RAYS * 3 * 4);
    float*          bgT  = (float*)carve(N_RAYS * 4);
    float*          bgSdf= (float*)carve((size_t)MB * 4);
    float*          bgRgb= (float*)carve((size_t)MB * 3 * 4);
    // bf16 weights (K padded to mult of 64, N to mult of 64)
    unsigned short* wiw0  = (unsigned short*)carve(64  * 256 * 2);
    unsigned short* wiw1  = (unsigned short*)carve(256 * 256 * 2);
    unsigned short* wiw2  = (unsigned short*)carve(256 * 256 * 2);
    unsigned short* wiw3  = (unsigned short*)carve(256 * 320 * 2);
    unsigned short* wiw1T = (unsigned short*)carve(256 * 256 * 2);
    unsigned short* wiw2T = (unsigned short*)carve(256 * 256 * 2);
    unsigned short* wrw0  = (unsigned short*)carve(320 * 256 * 2);
    unsigned short* wrw1  = (unsigned short*)carve(256 * 256 * 2);
    unsigned short* wbw0  = (unsigned short*)carve(64  * 256 * 2);
    unsigned short* wbw1  = (unsigned short*)carve(256 * 256 * 2);
    unsigned short* wbw2  = (unsigned short*)carve(256 * 320 * 2);
    unsigned short* wgw0  = (unsigned short*)carve(320 * 256 * 2);
    // bg big buffers reuse fg scratch (fg dead by then)
    unsigned short* Xbg  = X0;                      // [MB x 64]
    unsigned short* bgA  = bufA;                    // [MB x 256]
    unsigned short* bgB  = S1;                      // [MB x 256]
    unsigned short* bgF  = S2;                      // [MB x 256] (bg feat)
    unsigned short* Xg   = S3;                      // [MB x 320] (fits: 50MB buf)

    auto padw = [&](const float* s, int K, int N, unsigned short* dst, int Kp, int Np, int tr) {
        size_t tot = (size_t)Kp * Np;
        pad_weight_kernel<<<dim3((unsigned)((tot + 255) / 256)), 256, 0, stream>>>(s, K, N, dst, Kp, Np, tr);
    };
    padw(iw0, 3, 256,   wiw0, 64, 256, 0);
    padw(iw1, 256, 256, wiw1, 256, 256, 0);
    padw(iw2, 256, 256, wiw2, 256, 256, 0);
    padw(iw3, 256, 257, wiw3, 256, 320, 0);
    padw(iw1, 256, 256, wiw1T, 256, 256, 1);
    padw(iw2, 256, 256, wiw2T, 256, 256, 1);
    padw(rw0, 265, 256, wrw0, 320, 256, 0);
    padw(rw1, 256, 256, wrw1, 256, 256, 0);
    padw(bw0, 4, 256,   wbw0, 64, 256, 0);
    padw(bw1, 256, 256, wbw1, 256, 256, 0);
    padw(bw2, 256, 257, wbw2, 256, 320, 0);
    padw(gw0, 259, 256, wgw0, 320, 256, 0);

    auto gemm = [&](int cfg, const unsigned short* A, int lda, const unsigned short* B, int ldb,
                    const float* bias, int biasN, int Mm, int K,
                    unsigned short* C, unsigned short* Sig, const unsigned short* Mul,
                    float* sdfO, unsigned short* featO, int Np) {
        dim3 g(Mm / 128, Np / 64);
        switch (cfg) {
        case CFG_SP_SIG:
            gemm_bf16_kernel<1, false, true,  false><<<g, 256, 0, stream>>>(A, lda, B, ldb, bias, biasN, Mm, K, C, 256, Sig, Mul, sdfO, featO); break;
        case CFG_SPLIT:
            gemm_bf16_kernel<0, true,  false, false><<<g, 256, 0, stream>>>(A, lda, B, ldb, bias, biasN, Mm, K, C, 256, Sig, Mul, sdfO, featO); break;
        case CFG_MUL:
            gemm_bf16_kernel<0, false, false, true ><<<g, 256, 0, stream>>>(A, lda, B, ldb, bias, biasN, Mm, K, C, 256, Sig, Mul, sdfO, featO); break;
        case CFG_RELU:
            gemm_bf16_kernel<2, false, false, false><<<g, 256, 0, stream>>>(A, lda, B, ldb, bias, biasN, Mm, K, C, 256, Sig, Mul, sdfO, featO); break;
        case CFG_SP:
            gemm_bf16_kernel<1, false, false, false><<<g, 256, 0, stream>>>(A, lda, B, ldb, bias, biasN, Mm, K, C, 256, Sig, Mul, sdfO, featO); break;
        }
    };

    // ---- foreground points
    fg_pts_kernel<<<dim3((M + 255) / 256), 256, 0, stream>>>(cam, dirs, z1, z2, z3, X0, pts);

    // ---- implicit MLP forward (softplus100, store sigmoid masks)
    gemm(CFG_SP_SIG, X0, 64,    wiw0, 256, ib0, 256, M, 64,  bufA, S1, nullptr, nullptr, nullptr, 256);
    gemm(CFG_SP_SIG, bufA, 256, wiw1, 256, ib1, 256, M, 256, bufB, S2, nullptr, nullptr, nullptr, 256);
    gemm(CFG_SP_SIG, bufB, 256, wiw2, 256, ib2, 256, M, 256, bufA, S3, nullptr, nullptr, nullptr, 256);
    // layer3 split: sdf + feat
    gemm(CFG_SPLIT, bufA, 256, wiw3, 320, ib3, 257, M, 256, nullptr, nullptr, nullptr, sdf, bufB, 320);

    // ---- input gradient (VJP chain)
    g3_kernel<<<dim3((unsigned)(((size_t)M * 256 + 255) / 256)), 256, 0, stream>>>(S3, iw3, bufA);
    gemm(CFG_MUL, bufA, 256, wiw2T, 256, nullptr, 0, M, 256, S3, nullptr, S2, nullptr, nullptr, 256);
    gemm(CFG_MUL, S3,   256, wiw1T, 256, nullptr, 0, M, 256, S2, nullptr, S1, nullptr, nullptr, 256);
    gemm_smalln_kernel<<<dim3(M / 256), 256, 0, stream>>>(S2, 256, iw0, 256, 1, nullptr, M, 256, 3, 0, grads);

    // ---- rendering MLP
    assemble_xr_kernel<<<dim3((unsigned)(((size_t)M * 320 + 255) / 256)), 256, 0, stream>>>(pts, dirs, grads, bufB, Xr);
    gemm(CFG_RELU, Xr, 320,  wrw0, 256, rb0, 256, M, 320, bufA, nullptr, nullptr, nullptr, nullptr, 256);
    gemm(CFG_RELU, bufA, 256, wrw1, 256, rb1, 256, M, 256, bufB, nullptr, nullptr, nullptr, nullptr, 256);
    gemm_smalln_kernel<<<dim3(M / 256), 256, 0, stream>>>(bufB, 256, rw2, 3, 0, rb2, M, 256, 3, 3, rgb);

    // ---- foreground composite (long & short, mask select)
    composite_fg_kernel<<<dim3(2), 256, 0, stream>>>(z1, z2, z3, mask, sdf, rgb, fgRgb, bgT);

    // ---- background
    bg_pts_kernel<<<dim3((MB + 255) / 256), 256, 0, stream>>>(cam, dirs, zbg, Xbg);
    gemm(CFG_SP, Xbg, 64,  wbw0, 256, bb0, 256, MB, 64,  bgA, nullptr, nullptr, nullptr, nullptr, 256);
    gemm(CFG_SP, bgA, 256, wbw1, 256, bb1, 256, MB, 256, bgB, nullptr, nullptr, nullptr, nullptr, 256);
    gemm(CFG_SPLIT, bgB, 256, wbw2, 320, bb2, 257, MB, 256, nullptr, nullptr, nullptr, bgSdf, bgF, 320);
    assemble_xg_kernel<<<dim3((unsigned)(((size_t)MB * 320 + 255) / 256)), 256, 0, stream>>>(dirs, bgF, Xg);
    gemm(CFG_RELU, Xg, 320,  wgw0, 256, gb0, 256, MB, 320, bgA, nullptr, nullptr, nullptr, nullptr, 256);
    gemm_smalln_kernel<<<dim3(MB / 256), 256, 0, stream>>>(bgA, 256, gw1, 3, 0, gb1, MB, 256, 3, 3, bgRgb);

    // ---- background composite + final output
    composite_bg_kernel<<<dim3(2), 256, 0, stream>>>(zbg, bgSdf, bgRgb, fgRgb, bgT, outp);
}
